// Nonlocal_35287451304092
// MI455X (gfx1250) — matrix operand order repaired and checked by reading, not compiled, not run on hardware
//
#include <hip/hip_runtime.h>

typedef __attribute__((ext_vector_type(16))) __bf16 v16bf;
typedef __attribute__((ext_vector_type(8)))  float  v8f;
typedef int v4i __attribute__((vector_size(16)));  // matches builtin prototype

// Problem constants (from reference setup_inputs)
static constexpr int BB  = 2;
static constexpr int CC  = 512;
static constexpr int ICN = 256;
static constexpr int HH  = 64;
static constexpr int WWD = 64;
static constexpr int HW  = HH * WWD;      // 4096
static constexpr int K9  = ICN * 9;       // 2304

// ---------------------------------------------------------------------------
// CDNA5 async global->LDS copy (ASYNCcnt-tracked). Fallback: sync VGPR copy.
// ---------------------------------------------------------------------------
#if defined(__HIP_DEVICE_COMPILE__) && __has_builtin(__builtin_amdgcn_global_load_async_to_lds_b128)
#define HAVE_ASYNC_LDS 1
#else
#define HAVE_ASYNC_LDS 0
#endif

#define AS1 __attribute__((address_space(1)))
#define AS3 __attribute__((address_space(3)))

__device__ __forceinline__ void async_copy16(const __bf16* g, __bf16* l) {
#if HAVE_ASYNC_LDS
  __builtin_amdgcn_global_load_async_to_lds_b128((AS1 v4i*)g, (AS3 v4i*)l, 0, 0);
#else
  *reinterpret_cast<float4*>(l) = *reinterpret_cast<const float4*>(g);
#endif
}

template <int N>
__device__ __forceinline__ void wait_async() {
#if HAVE_ASYNC_LDS
#if __has_builtin(__builtin_amdgcn_s_wait_asynccnt)
  __builtin_amdgcn_s_wait_asynccnt(N);
#else
  asm volatile("s_wait_asynccnt %0" ::"i"(N));
#endif
#endif
}

// ---------------------------------------------------------------------------
// WMMA helpers (wave32; A/B k order chosen here, C/D per ISA)
// A 16x32 bf16 (code's k order): lane<16 -> row=lane, K contiguous [0..15]; lane>=16 -> row=lane-16, K [16..31]
// B 32x16 bf16 (code's k order): lane<16 -> col=lane, K contiguous [0..15]; lane>=16 -> col=lane-16, K [16..31]
// C/D 16x16 f32: VGPR i: lanes0-15 M=i, lanes16-31 M=i+8; N = lane&15
// ---------------------------------------------------------------------------
__device__ __forceinline__ v16bf frag_a(const __bf16* base, int ld, int lane) {
  const int half = lane >> 4, l = lane & 15;
  const __bf16* q = base + (size_t)l * ld + (half << 4);
  v16bf r;
  reinterpret_cast<float4*>(&r)[0] = *reinterpret_cast<const float4*>(q);
  reinterpret_cast<float4*>(&r)[1] = *reinterpret_cast<const float4*>(q + 8);
  return r;
}
__device__ __forceinline__ v16bf frag_b(const __bf16* base, int ld, int lane) {
  const int half = lane >> 4, l = lane & 15;
  const __bf16* q = base + (size_t)l * ld + (half << 4);
  v16bf r;
  reinterpret_cast<float4*>(&r)[0] = reinterpret_cast<const float4*>(q)[0];
  reinterpret_cast<float4*>(&r)[1] = reinterpret_cast<const float4*>(q)[1];
  return r;
}
__device__ __forceinline__ v8f wmma_bf16(v16bf a, v16bf b, v8f c) {
  return __builtin_amdgcn_wmma_f32_16x16x32_bf16(false, a, false, b, (short)0, c, false, false);
}

// ---------------------------------------------------------------------------
// 1) f32 -> bf16 convert (weights)
// ---------------------------------------------------------------------------
__global__ __launch_bounds__(256) void cvt_bf16_kernel(const float* __restrict__ in,
                                                       __bf16* __restrict__ out, int n) {
  int i = blockIdx.x * 256 + threadIdx.x;
  if (i < n) out[i] = (__bf16)in[i];
}

// ---------------------------------------------------------------------------
// 2) instance-norm stats: one block per (b,c), reduce HW elements
// ---------------------------------------------------------------------------
__global__ __launch_bounds__(256) void stats_kernel(const float* __restrict__ X,
                                                    float* __restrict__ mean_out,
                                                    float* __restrict__ rstd_out) {
  const float* x = X + (size_t)blockIdx.x * HW;
  float s = 0.f, s2 = 0.f;
  for (int i = threadIdx.x; i < HW; i += 256) { float v = x[i]; s += v; s2 += v * v; }
  #pragma unroll
  for (int off = 16; off >= 1; off >>= 1) {
    s  += __shfl_xor(s,  off, 32);
    s2 += __shfl_xor(s2, off, 32);
  }
  __shared__ float sh_s[8], sh_s2[8];
  int warp = threadIdx.x >> 5, lane = threadIdx.x & 31;
  if (lane == 0) { sh_s[warp] = s; sh_s2[warp] = s2; }
  __syncthreads();
  if (threadIdx.x == 0) {
    float ts = 0.f, ts2 = 0.f;
    #pragma unroll
    for (int w = 0; w < 8; w++) { ts += sh_s[w]; ts2 += sh_s2[w]; }
    float m   = ts / (float)HW;
    float var = (ts2 - (float)HW * m * m) / (float)(HW - 1);
    mean_out[blockIdx.x] = m;
    rstd_out[blockIdx.x] = rsqrtf(var + 1e-5f);
  }
}

// ---------------------------------------------------------------------------
// 3) normalize (optional) + transpose: (B,C,HW) f32 -> (B,HW,C) bf16
// ---------------------------------------------------------------------------
__global__ __launch_bounds__(256) void norm_t_kernel(const float* __restrict__ X,
                                                     const float* __restrict__ mean,
                                                     const float* __restrict__ rstd,
                                                     __bf16* __restrict__ Out) {
  size_t idx = (size_t)blockIdx.x * 256 + threadIdx.x;  // total = BB*CC*HW
  int n = (int)(idx % HW);
  int c = (int)((idx / HW) % CC);
  int b = (int)(idx / ((size_t)HW * CC));
  float v = X[idx];
  if (mean) v = (v - mean[b * CC + c]) * rstd[b * CC + c];
  Out[((size_t)b * HW + n) * CC + c] = (__bf16)v;
}

// ---------------------------------------------------------------------------
// 4) Generic WMMA GEMM: D[b,m,n] = sum_k A[m,k]*Bt[b,n,k] + bias[m]
//    A:(M,K) bf16, Bt:(BATCH,N,K) bf16, D:(BATCH,M,N) OutT
//    Block = 4 waves, tile 64(M) x 64(N). grid = (N/64, M/64, BATCH)
// ---------------------------------------------------------------------------
template <typename OutT>
__global__ __launch_bounds__(128) void gemm_kernel(const __bf16* __restrict__ A,
                                                   const __bf16* __restrict__ Bt,
                                                   const float* __restrict__ bias,
                                                   OutT* __restrict__ Dst,
                                                   int M, int N, int K) {
  const int wave = threadIdx.x >> 5, lane = threadIdx.x & 31;
  const int half = lane >> 4, l = lane & 15;
  const int m0 = blockIdx.y * 64 + wave * 16;
  const int n0 = blockIdx.x * 64;
  const __bf16* Ab = A + (size_t)m0 * K;
  const __bf16* Bb = Bt + ((size_t)blockIdx.z * N + n0) * K;
  v8f acc[4] = {};
  for (int k = 0; k < K; k += 32) {
    v16bf a = frag_a(Ab + k, K, lane);
    #pragma unroll
    for (int t = 0; t < 4; t++) {
      v16bf bb = frag_b(Bb + (size_t)(t * 16) * K + k, K, lane);
      acc[t] = wmma_bf16(a, bb, acc[t]);
    }
  }
  size_t dstBase = (size_t)blockIdx.z * M * N;
  #pragma unroll
  for (int t = 0; t < 4; t++) {
    #pragma unroll
    for (int i = 0; i < 8; i++) {
      int row = m0 + i + half * 8;
      int col = n0 + t * 16 + l;
      float v = acc[t][i] + (bias ? bias[row] : 0.f);
      Dst[dstBase + (size_t)row * N + col] = (OutT)v;
    }
  }
}

// ---------------------------------------------------------------------------
// 5) patch unfold (reflect pad 1, 3x3) + scale:
//    (B,IC,HW) bf16 -> (B,HW,IC*9) bf16, torch layout (c slow, k fast)
// ---------------------------------------------------------------------------
__global__ __launch_bounds__(256) void patches_kernel(const __bf16* __restrict__ X,
                                                      const float* __restrict__ scale,
                                                      __bf16* __restrict__ P) {
  size_t idx = (size_t)blockIdx.x * 256 + threadIdx.x;  // total = BB*HW*ICN
  int c = (int)(idx % ICN);
  int lidx = (int)((idx / ICN) % HW);
  int b = (int)(idx / ((size_t)ICN * HW));
  int i = lidx / WWD, j = lidx % WWD;
  const __bf16* x = X + ((size_t)b * ICN + c) * HW;
  __bf16* out = P + ((size_t)b * HW + lidx) * K9 + c * 9;
  #pragma unroll
  for (int di = 0; di < 3; di++) {
    int r = i + di - 1;
    r = (r < 0) ? 1 : ((r >= HH) ? (2 * HH - 2 - r) : r);
    #pragma unroll
    for (int dj = 0; dj < 3; dj++) {
      int cc = j + dj - 1;
      cc = (cc < 0) ? 1 : ((cc >= WWD) ? (2 * WWD - 2 - cc) : cc);
      out[di * 3 + dj] = (__bf16)((float)x[r * WWD + cc] * scale[di * 3 + dj]);
    }
  }
}

// ---------------------------------------------------------------------------
// 6) Flash attention: y[b,l,ic] = softmax_m( theta_p[l,:]·phi_p[m,:] ) · g[ic,m]
//    Block = 4 waves; 64 query rows per block, 64-key tiles, K=2304 score loop.
//    phi chunks double-buffered in LDS via async global->LDS copies (ASYNCcnt);
//    LDS rows padded (+8 elems) to avoid 16-way bank conflicts on B-frag reads.
//    g B-fragments read straight from (IC,HW) global (L2-resident).
// ---------------------------------------------------------------------------
static constexpr int PHI_LD = 136;  // 128 + 8
static constexpr int P_LD   = 72;   // 64 + 8

__device__ __forceinline__ void stage_phi_chunk(const __bf16* __restrict__ src,  // (row0, k0)
                                                __bf16* dstBuf, int tid) {
  // 64 rows x 128 K; 128 threads -> each thread half a row = 8 x 16B segments
  const int r = tid >> 1, part = tid & 1;
  const __bf16* g = src + (size_t)r * K9 + part * 64;
  __bf16* l = dstBuf + r * PHI_LD + part * 64;
  #pragma unroll
  for (int q = 0; q < 8; q++) async_copy16(g + q * 8, l + q * 8);
}

__global__ __launch_bounds__(128) void attention_kernel(const __bf16* __restrict__ ThetaP,
                                                        const __bf16* __restrict__ PhiP,
                                                        const __bf16* __restrict__ G,
                                                        __bf16* __restrict__ Y) {
  __shared__ __align__(16) __bf16 sPhi[2][64 * PHI_LD];  // 2 x 17408 B
  __shared__ __align__(16) __bf16 sP[4][16 * P_LD];      //     9216 B

  const int b = blockIdx.y;
  const int m_base = blockIdx.x * 64;
  const int tid = threadIdx.x;
  const int wave = tid >> 5, lane = tid & 31;
  const int half = lane >> 4, l = lane & 15;

  const __bf16* thetaWave = ThetaP + ((size_t)b * HW + m_base + wave * 16) * K9;
  const __bf16* phiBase = PhiP + (size_t)b * HW * K9;
  const __bf16* gB = G + (size_t)b * ICN * HW;

  float m_run[8], l_run[8];
  #pragma unroll
  for (int i = 0; i < 8; i++) { m_run[i] = -3.0e38f; l_run[i] = 0.f; }
  v8f yacc[16] = {};

  int buf = 0;
  // prologue: kick chunk (n0=0, k0=0) into buffer 0
  stage_phi_chunk(phiBase, sPhi[0], tid);

  for (int n0 = 0; n0 < HW; n0 += 64) {
    v8f sacc[4] = {};
    for (int k0 = 0; k0 < K9; k0 += 128) {
      __syncthreads();  // everyone done computing on the buffer we refill next
      int k1 = k0 + 128, n1 = n0;
      if (k1 == K9) { k1 = 0; n1 = n0 + 64; }
      const bool hasNext = (n1 < HW);
      if (hasNext)
        stage_phi_chunk(phiBase + (size_t)n1 * K9 + k1, sPhi[buf ^ 1], tid);
      if (hasNext) wait_async<8>(); else wait_async<0>();
      __syncthreads();  // current chunk landed for all threads
      const __bf16* cur = sPhi[buf];
      #pragma unroll
      for (int kk = 0; kk < 128; kk += 32) {
        v16bf a = frag_a(thetaWave + k0 + kk, K9, lane);  // global (L2-resident)
        #pragma unroll
        for (int t = 0; t < 4; t++) {
          v16bf bb = frag_b(cur + t * 16 * PHI_LD + kk, PHI_LD, lane);
          sacc[t] = wmma_bf16(a, bb, sacc[t]);
        }
      }
      buf ^= 1;
    }

    // ----- online softmax over this 64-key tile -----
    __bf16* sPw = sP[wave];
    float corr[8];
    #pragma unroll
    for (int i = 0; i < 8; i++) {
      float v = fmaxf(fmaxf(sacc[0][i], sacc[1][i]), fmaxf(sacc[2][i], sacc[3][i]));
      #pragma unroll
      for (int off = 8; off >= 1; off >>= 1) v = fmaxf(v, __shfl_xor(v, off, 32));
      float mn = fmaxf(m_run[i], v);
      float c  = __expf(m_run[i] - mn);
      m_run[i] = mn;
      corr[i]  = c;
      float rs = 0.f;
      #pragma unroll
      for (int t = 0; t < 4; t++) {
        float p = __expf(sacc[t][i] - mn);
        rs += p;
        sPw[(i + half * 8) * P_LD + t * 16 + l] = (__bf16)p;
      }
      #pragma unroll
      for (int off = 8; off >= 1; off >>= 1) rs += __shfl_xor(rs, off, 32);
      l_run[i] = l_run[i] * c + rs;
    }
    #pragma unroll
    for (int t = 0; t < 16; t++)
      #pragma unroll
      for (int i = 0; i < 8; i++) yacc[t][i] *= corr[i];

    __syncthreads();  // P stores visible (forces DS waits)

    // ----- y += P (16x64) @ g (64x256), g B-frags straight from global -----
    const __bf16* gTile = gB + n0;  // row ic stride = HW
    #pragma unroll
    for (int ks = 0; ks < 2; ks++) {
      v16bf a = frag_a(sPw + ks * 32, P_LD, lane);
      #pragma unroll
      for (int t = 0; t < 16; t++) {
        v16bf bb = frag_b(gTile + (size_t)(t * 16) * HW + ks * 32, HW, lane);
        yacc[t] = wmma_bf16(a, bb, yacc[t]);
      }
    }
  }

  // normalize by running sum; store y (B,HW,IC) bf16
  float inv[8];
  #pragma unroll
  for (int i = 0; i < 8; i++) inv[i] = (l_run[i] > 0.f) ? (1.f / l_run[i]) : 0.f;
  __bf16* yRow = Y + ((size_t)b * HW + m_base + wave * 16) * ICN;
  #pragma unroll
  for (int t = 0; t < 16; t++)
    #pragma unroll
    for (int i = 0; i < 8; i++) {
      int row = i + half * 8;
      yRow[(size_t)row * ICN + t * 16 + l] = (__bf16)(yacc[t][i] * inv[i]);
    }
}

// ---------------------------------------------------------------------------
// host launcher
// ---------------------------------------------------------------------------
extern "C" void kernel_launch(void* const* d_in, const int* in_sizes, int n_in,
                              void* d_out, int out_size, void* d_ws, size_t ws_size,
                              hipStream_t stream) {
  (void)in_sizes; (void)n_in; (void)out_size; (void)ws_size;
  const float* content  = (const float*)d_in[0];
  const float* style    = (const float*)d_in[1];
  const float* fusion   = (const float*)d_in[2];
  const float* theta_w  = (const float*)d_in[3];
  const float* theta_b  = (const float*)d_in[4];
  const float* phi_w    = (const float*)d_in[5];
  const float* phi_b    = (const float*)d_in[6];
  const float* g_w      = (const float*)d_in[7];
  const float* g_b      = (const float*)d_in[8];
  const float* W_w      = (const float*)d_in[9];
  const float* W_b      = (const float*)d_in[10];
  const float* scale    = (const float*)d_in[11];
  float* out = (float*)d_out;

  char* ws = (char*)d_ws;
  size_t off = 0;
  auto alloc = [&](size_t bytes) -> char* {
    char* p = ws + off;
    off += (bytes + 255) & ~(size_t)255;
    return p;
  };
  __bf16* theta_wb = (__bf16*)alloc((size_t)ICN * CC * 2);
  __bf16* phi_wb   = (__bf16*)alloc((size_t)ICN * CC * 2);
  __bf16* g_wb     = (__bf16*)alloc((size_t)ICN * CC * 2);
  __bf16* W_wb     = (__bf16*)alloc((size_t)CC * ICN * 2);
  float*  mean_c   = (float*)alloc((size_t)BB * CC * 4);
  float*  rstd_c   = (float*)alloc((size_t)BB * CC * 4);
  float*  mean_s   = (float*)alloc((size_t)BB * CC * 4);
  float*  rstd_s   = (float*)alloc((size_t)BB * CC * 4);
  __bf16* cnt_nt   = (__bf16*)alloc((size_t)BB * HW * CC * 2);
  __bf16* sty_nt   = (__bf16*)alloc((size_t)BB * HW * CC * 2);
  __bf16* fus_nt   = (__bf16*)alloc((size_t)BB * HW * CC * 2);
  __bf16* theta_x  = (__bf16*)alloc((size_t)BB * ICN * HW * 2);
  __bf16* phi_sx   = (__bf16*)alloc((size_t)BB * ICN * HW * 2);
  __bf16* g_sx     = (__bf16*)alloc((size_t)BB * ICN * HW * 2);
  __bf16* theta_p  = (__bf16*)alloc((size_t)BB * HW * K9 * 2);
  __bf16* phi_p    = (__bf16*)alloc((size_t)BB * HW * K9 * 2);
  __bf16* yb       = (__bf16*)alloc((size_t)BB * HW * ICN * 2);

  // 1) weights -> bf16
  {
    int n = ICN * CC;
    cvt_bf16_kernel<<<(n + 255) / 256, 256, 0, stream>>>(theta_w, theta_wb, n);
    cvt_bf16_kernel<<<(n + 255) / 256, 256, 0, stream>>>(phi_w,   phi_wb,   n);
    cvt_bf16_kernel<<<(n + 255) / 256, 256, 0, stream>>>(g_w,     g_wb,     n);
    cvt_bf16_kernel<<<(n + 255) / 256, 256, 0, stream>>>(W_w,     W_wb,     n);
  }
  // 2) instance-norm stats
  stats_kernel<<<BB * CC, 256, 0, stream>>>(content, mean_c, rstd_c);
  stats_kernel<<<BB * CC, 256, 0, stream>>>(style,   mean_s, rstd_s);
  // 3) normalize + transpose to (B,HW,C) bf16
  {
    int blocks = (int)(((size_t)BB * CC * HW) / 256);
    norm_t_kernel<<<blocks, 256, 0, stream>>>(content, mean_c, rstd_c, cnt_nt);
    norm_t_kernel<<<blocks, 256, 0, stream>>>(style,   mean_s, rstd_s, sty_nt);
    norm_t_kernel<<<blocks, 256, 0, stream>>>(fusion,  nullptr, nullptr, fus_nt);
  }
  // 4) 1x1 convs via WMMA GEMM -> (B,IC,HW) bf16
  {
    dim3 grid(HW / 64, ICN / 64, BB);
    gemm_kernel<__bf16><<<grid, 128, 0, stream>>>(theta_wb, cnt_nt, theta_b, theta_x, ICN, HW, CC);
    gemm_kernel<__bf16><<<grid, 128, 0, stream>>>(phi_wb,   sty_nt, phi_b,   phi_sx, ICN, HW, CC);
    gemm_kernel<__bf16><<<grid, 128, 0, stream>>>(g_wb,     fus_nt, g_b,     g_sx,   ICN, HW, CC);
  }
  // 5) patch unfold + scale
  {
    int blocks = (int)(((size_t)BB * HW * ICN) / 256);
    patches_kernel<<<blocks, 256, 0, stream>>>(theta_x, scale, theta_p);
    patches_kernel<<<blocks, 256, 0, stream>>>(phi_sx,  scale, phi_p);
  }
  // 6) flash attention -> y (B,HW,IC) bf16
  attention_kernel<<<dim3(HW / 64, BB), 128, 0, stream>>>(theta_p, phi_p, g_sx, yb);
  // 7) final 1x1 conv W_w @ y -> out (B,C,H,W) f32
  {
    dim3 grid(HW / 64, CC / 64, BB);
    gemm_kernel<float><<<grid, 128, 0, stream>>>(W_wb, yb, W_b, out, CC, HW, ICN);
  }
}
